// AConv_38319698215157
// MI455X (gfx1250) — compile-verified, matrix operand order checked by reading, not run on hardware
//
#include <hip/hip_runtime.h>

#define B_   4
#define C_   64
#define H_   192
#define W_   640
#define C25  25
#define HW_  (H_*W_)

typedef __attribute__((ext_vector_type(16))) _Float16 v16h;
typedef __attribute__((ext_vector_type(8)))  _Float16 v8h;
typedef __attribute__((ext_vector_type(8)))  float    v8f;
typedef __attribute__((ext_vector_type(4)))  unsigned int v4u;
typedef __attribute__((ext_vector_type(8)))  int      v8i;
typedef __attribute__((ext_vector_type(4)))  int      v4i;

// ---------------------------------------------------------------------------
// Instance-norm statistics: one block per (b, c); writes mean and rsqrt(var+eps)
// ---------------------------------------------------------------------------
__global__ __launch_bounds__(256)
void in_stats_kernel(const float* __restrict__ y, float* __restrict__ mu,
                     float* __restrict__ rs) {
  int bc = blockIdx.x;
  const float* p = y + (size_t)bc*HW_;
  float s = 0.f, ss = 0.f;
  for (int i = threadIdx.x; i < HW_; i += 256) {
    float v = p[i];
    s += v; ss += v*v;
  }
  __shared__ float sh1[256], sh2[256];
  sh1[threadIdx.x] = s; sh2[threadIdx.x] = ss;
  __syncthreads();
  for (int off = 128; off > 0; off >>= 1) {
    if (threadIdx.x < off) {
      sh1[threadIdx.x] += sh1[threadIdx.x + off];
      sh2[threadIdx.x] += sh2[threadIdx.x + off];
    }
    __syncthreads();
  }
  if (threadIdx.x == 0) {
    float m = sh1[0] / (float)HW_;
    float v = sh2[0] / (float)HW_ - m*m;
    mu[bc] = m;
    rs[bc] = rsqrtf(v + 1e-5f);
  }
}

// ---------------------------------------------------------------------------
// Generic 3x3-conv weight packer -> per-lane WMMA A-fragment order (f16).
// Fragment f = (tap*KS + s)*2 + mt  (2 M-tiles of 16 for Cout=25, zero-padded).
// elem idx = f*512 + lane*16 + e ; kmap(g,e) = 8g+e (e<8) ; 16+8g+(e-8) (e>=8)
// ---------------------------------------------------------------------------
__global__ __launch_bounds__(256)
void pack_conv_kernel(const float* __restrict__ w, _Float16* __restrict__ out,
                      int Cin, int KS, int nfrag) {
  int idx = blockIdx.x*256 + threadIdx.x;
  if (idx >= nfrag*512) return;
  int f    = idx >> 9;
  int rloc = idx & 511;
  int lane = rloc >> 4;
  int e    = rloc & 15;
  int mt   = f & 1;
  int f2   = f >> 1;
  int s    = f2 % KS;
  int tap  = f2 / KS;
  int g    = lane >> 4;
  int m    = lane & 15;
  int k    = 8*g + e + ((e >= 8) ? 8 : 0);
  int c    = 32*s + k;
  int oc   = mt*16 + m;
  float v  = (oc < C25 && c < Cin) ? w[(oc*Cin + c)*9 + tap] : 0.f;
  out[idx] = (_Float16)v;
}

// ---------------------------------------------------------------------------
// 3x3 conv on WMMA, im2col with tap-major K (k = tap*Cin + ci) so B-fragment
// channel runs are contiguous in the [row][w][c] LDS tile.
// Each block: (b, h, 64-pixel segment), 8 waves; wave (mt, nt) computes a
// 16-out-channel x 16-pixel tile via 9*KS v_wmma_f32_16x16x32_f16.
// CSTG: staged channels (64 or 25->32 zero-padded); NORM fuses instance norm.
// ---------------------------------------------------------------------------
template<int CSTG, int CIN_R, int KS, bool NORM, bool RELU, bool BIAS>
__global__ __launch_bounds__(256)
void conv3x3_wmma_kernel(const float* __restrict__ in, const _Float16* __restrict__ wf,
                         const float* __restrict__ mu, const float* __restrict__ rs,
                         const float* __restrict__ bias, float* __restrict__ out) {
  __shared__ __align__(16) _Float16 xs2[3][66][CSTG];   // [row][w][c], c fastest

  int tid  = threadIdx.x;
  int blk  = blockIdx.x;
  int wseg = blk % 10;
  int h    = (blk / 10) % H_;
  int b    = blk / (10*H_);
  int w0   = wseg*64;

  const float* ib = in + (size_t)b*CIN_R*HW_;
  for (int t = tid; t < 3*66*CSTG; t += 256) {
    int ww = t % 66;
    int r  = (t / 66) % 3;
    int c  = t / (66*3);
    int hg = h + r - 1;
    int wg = w0 + ww - 1;
    float v = 0.f;
    if (c < CIN_R && hg >= 0 && hg < H_ && wg >= 0 && wg < W_) {
      v = ib[(size_t)c*HW_ + hg*W_ + wg];
      if (NORM) v = (v - mu[b*CIN_R + c]) * rs[b*CIN_R + c];
    }
    xs2[r][ww][c] = (_Float16)v;   // pad-after-norm => OOB taps are exact zeros
  }
  __syncthreads();

  int wave = tid >> 5;
  int lane = tid & 31;
  int mt   = wave & 1;         // out-channel tile (x16), Cout padded 25->32
  int nt   = wave >> 1;        // pixel tile (x16)
  int g    = lane >> 4;
  int ln   = lane & 15;
  int pix  = nt*16 + ln;

  v8f acc = {};
  #pragma unroll
  for (int tap = 0; tap < 9; ++tap) {
    int i = tap/3, j = tap%3;
    const _Float16* xrow = &xs2[i][pix + j][0];
    #pragma unroll
    for (int s = 0; s < KS; ++s) {
      v16h a = *(const v16h*)(wf + (((size_t)(tap*KS + s)*2 + mt) << 9) + lane*16);
      v8h lo = *(const v8h*)(xrow + 32*s + 8*g);
      v8h hi = *(const v8h*)(xrow + 32*s + 16 + 8*g);
      v16h bv;
      #pragma unroll
      for (int e = 0; e < 8; ++e) { bv[e] = lo[e]; bv[e+8] = hi[e]; }
      acc = __builtin_amdgcn_wmma_f32_16x16x32_f16(
          false, a, false, bv, (short)0, acc, false, false);
    }
  }

  int o0 = mt*16;
  float* ob = out + (size_t)b*C25*HW_ + (size_t)h*W_ + w0 + pix;
  #pragma unroll
  for (int r2 = 0; r2 < 8; ++r2) {
    int oc = o0 + r2 + 8*g;              // D layout: M = r + 8*g, N = lane%16
    if (oc < C25) {
      float v = acc[r2];
      if (BIAS) v += bias[oc];
      if (RELU) v = fmaxf(v, 0.f);
      ob[(size_t)oc*HW_] = v;
    }
  }
}

// ---------------------------------------------------------------------------
// Pre-pack pac_w (Cout=64 -> 4 M-tiles) into WMMA A-fragment order (f16).
// Fragment f = ((tap*2 + s)*4 + mt), elem idx = f*512 + lane*16 + e.
// ---------------------------------------------------------------------------
__global__ __launch_bounds__(256)
void pack_pacw_kernel(const float* __restrict__ pw, _Float16* __restrict__ out) {
  int idx  = blockIdx.x*256 + threadIdx.x;    // < 200*512
  int f    = idx >> 9;
  int rloc = idx & 511;
  int lane = rloc >> 4;
  int e    = rloc & 15;
  int mt   = f & 3;
  int s    = (f >> 2) & 1;
  int tap  = f >> 3;
  int g    = lane >> 4;
  int m    = lane & 15;
  int k    = 8*g + e + ((e >= 8) ? 8 : 0);
  int oc   = mt*16 + m;
  int c    = 32*s + k;
  out[idx] = (_Float16)pw[(oc*C_ + c)*25 + tap];   // pac_w[oc][c][i][j], tap=i*5+j
}

// ---------------------------------------------------------------------------
// PAC stage on WMMA: each block = (b, h, 64-pixel segment), 16 waves.
// K tile (25 x 64 f32, row stride H*W) is DMA'd into LDS by the Tensor Data
// Mover (2D D# descriptor, no halo / no OOB), overlapping the VALU staging of
// the x halo tile.  Wave (mt, nt) then computes 16 out-channels x 16 pixels
// via 25 taps x 2 ksteps of v_wmma_f32_16x16x32_f16.
// ---------------------------------------------------------------------------
__global__ __launch_bounds__(512)
void pac_wmma_kernel(const float* __restrict__ x, const float* __restrict__ Kb,
                     const _Float16* __restrict__ pwh, const float* __restrict__ pac_b,
                     float* __restrict__ out) {
  __shared__ __align__(16) _Float16 xs[5][68][C_];   // [row][w][c], c fastest
  __shared__ __align__(16) float ks32[C25][64];      // [tap][pixel]

  int tid  = threadIdx.x;
  int blk  = blockIdx.x;
  int wseg = blk % 10;
  int h    = (blk / 10) % H_;
  int b    = blk / (10*H_);
  int w0   = wseg*64;

  const float* kbp = Kb + (size_t)b*C25*HW_ + (size_t)h*W_ + w0;

#if defined(__has_builtin) && __has_builtin(__builtin_amdgcn_tensor_load_to_lds)
  if (tid < 32) {
    // D# group 0: count=1, lds_addr, 57-bit global addr, type=2 ("image")
    unsigned long long ga = (unsigned long long)(uintptr_t)kbp;
    unsigned lds_off = (unsigned)(uintptr_t)&ks32[0][0];
    v4u g0 = { 1u,                                    // count=1, user mode
               lds_off,
               (unsigned)(ga & 0xFFFFFFFFu),
               (unsigned)((ga >> 32) & 0x01FFFFFFu) | (2u << 30) };
    // D# group 1: data_size=4B; tensor 64x25 (dim0 contiguous), tile 64x25,
    // tensor_dim0_stride = H*W elements.
    v8i g1 = { (int)(2u << 16),                       // data_size = 4 bytes
               (int)(64u << 16),                      // tensor_dim0 = 64
               (int)(25u << 16),                      // tensor_dim1 = 25
               (int)(64u << 16),                      // tile_dim0   = 64
               (int)25,                               // tile_dim1   = 25, tile_dim2=0
               (int)HW_,                              // tensor_dim0_stride = 122880
               0, 0 };
    v4i gz4 = { 0, 0, 0, 0 };                         // groups 2/3 unused (2D)
    v8i gz8 = { 0, 0, 0, 0, 0, 0, 0, 0 };             // trailing group (unused)
    __builtin_amdgcn_tensor_load_to_lds(g0, g1, gz4, gz4, gz8, 0);
  }
#else
  for (int t = tid; t < C25*64; t += 512)
    ks32[t / 64][t % 64] = kbp[(size_t)(t / 64)*HW_ + (t % 64)];
#endif

  // stage x halo tile (5 rows x 68 cols x 64 ch) into LDS as f16 (overlaps TDM)
  const float* xb = x + (size_t)b*C_*HW_;
  for (int t = tid; t < 5*68*C_; t += 512) {
    int ww = t % 68;
    int r  = (t / 68) % 5;
    int c  = t / (68*5);
    int hg = h + r - 2;
    int wg = w0 + ww - 2;
    float v = (hg >= 0 && hg < H_ && wg >= 0 && wg < W_)
                  ? xb[(size_t)c*HW_ + hg*W_ + wg] : 0.f;
    xs[r][ww][c] = (_Float16)v;
  }

#if defined(__has_builtin) && __has_builtin(__builtin_amdgcn_tensor_load_to_lds)
  if (tid < 32) __builtin_amdgcn_s_wait_tensorcnt(0);
#endif
  __syncthreads();

  int wave = tid >> 5;
  int lane = tid & 31;
  int mt   = wave & 3;         // out-channel tile (x16)
  int nt   = wave >> 2;        // pixel tile (x16)
  int g    = lane >> 4;
  int ln   = lane & 15;
  int pix  = nt*16 + ln;       // this lane's pixel (B/D column)

  v8f acc = {};
  #pragma unroll 5
  for (int tap = 0; tap < 25; ++tap) {
    int i = tap / 5, j = tap % 5;
    const _Float16* xrow = &xs[i][pix + j][0];
    _Float16 kv = (_Float16)ks32[tap][pix];
    #pragma unroll
    for (int s = 0; s < 2; ++s) {
      // A-fragment: prepacked, one 32B load per lane (L2-resident)
      v16h a = *(const v16h*)(pwh + (((size_t)(tap*2 + s)*4 + mt) << 9) + lane*16);
      // B-fragment: two contiguous 8-channel runs from LDS, scaled by K scalar
      v8h lo = *(const v8h*)(xrow + 32*s + 8*g);
      v8h hi = *(const v8h*)(xrow + 32*s + 16 + 8*g);
      v16h bv;
      #pragma unroll
      for (int e = 0; e < 8; ++e) { bv[e] = lo[e]*kv; bv[e+8] = hi[e]*kv; }
      acc = __builtin_amdgcn_wmma_f32_16x16x32_f16(
          /*neg_a=*/false, a, /*neg_b=*/false, bv,
          /*c_mod=*/(short)0, acc, /*reuse_a=*/false, /*reuse_b=*/false);
    }
  }

  // epilogue: bias + relu + residual
  int o0 = mt*16;
  size_t base = (size_t)b*C_*HW_ + (size_t)h*W_ + w0 + pix;
  #pragma unroll
  for (int r2 = 0; r2 < 8; ++r2) {
    int oc  = o0 + r2 + 8*g;               // D layout: M = r + 8*g, N = lane%16
    float v = acc[r2] + pac_b[oc];
    v = fmaxf(v, 0.f);
    size_t idx = base + (size_t)oc*HW_;
    out[idx] = v + x[idx];
  }
}

// ---------------------------------------------------------------------------
extern "C" void kernel_launch(void* const* d_in, const int* in_sizes, int n_in,
                              void* d_out, int out_size, void* d_ws, size_t ws_size,
                              hipStream_t stream) {
  (void)in_sizes; (void)n_in; (void)out_size; (void)ws_size;
  const float* x    = (const float*)d_in[0];
  const float* w1   = (const float*)d_in[1];
  const float* w2   = (const float*)d_in[2];
  const float* w3   = (const float*)d_in[3];
  const float* b3   = (const float*)d_in[4];
  const float* pacw = (const float*)d_in[5];
  const float* pacb = (const float*)d_in[6];
  float* out = (float*)d_out;

  char* ws = (char*)d_ws;
  const size_t szPlane = (size_t)B_*C25*HW_*sizeof(float);   // 49,152,000 B
  float*    y1  = (float*)(ws);
  float*    y2  = (float*)(ws + szPlane);
  float*    Kb  = (float*)(ws + 2*szPlane);
  size_t off = 3*szPlane;
  _Float16* pwhP = (_Float16*)(ws + off); off += (size_t)200*512*sizeof(_Float16);
  _Float16* pwh1 = (_Float16*)(ws + off); off += (size_t)36*512*sizeof(_Float16);
  _Float16* pwh2 = (_Float16*)(ws + off); off += (size_t)18*512*sizeof(_Float16);
  _Float16* pwh3 = (_Float16*)(ws + off); off += (size_t)18*512*sizeof(_Float16);
  float* stats = (float*)(ws + off);
  float* mu1 = stats;       float* rs1 = stats + 128;
  float* mu2 = stats + 256; float* rs2 = stats + 384;

  const int segBlocks = B_*H_*10;         // 7680 (b, h, 64-px segment)

  // pack all weights into WMMA fragment layouts (tiny, L2-resident)
  pack_pacw_kernel<<<(200*512)/256, 256, 0, stream>>>(pacw, pwhP);
  pack_conv_kernel<<<(36*512)/256, 256, 0, stream>>>(w1, pwh1, C_,  2, 36);
  pack_conv_kernel<<<(18*512)/256, 256, 0, stream>>>(w2, pwh2, C25, 1, 18);
  pack_conv_kernel<<<(18*512)/256, 256, 0, stream>>>(w3, pwh3, C25, 1, 18);

  // conv1 (64->25) + relu, WMMA
  conv3x3_wmma_kernel<64, C_, 2, false, true, false>
      <<<segBlocks, 256, 0, stream>>>(x, pwh1, nullptr, nullptr, nullptr, y1);
  in_stats_kernel<<<B_*C25, 256, 0, stream>>>(y1, mu1, rs1);
  // conv2 (25->25) + relu, normalized input, WMMA
  conv3x3_wmma_kernel<32, C25, 1, true, true, false>
      <<<segBlocks, 256, 0, stream>>>(y1, pwh2, mu1, rs1, nullptr, y2);
  in_stats_kernel<<<B_*C25, 256, 0, stream>>>(y2, mu2, rs2);
  // conv3 (25->25) + bias, normalized input, WMMA
  conv3x3_wmma_kernel<32, C25, 1, true, false, true>
      <<<segBlocks, 256, 0, stream>>>(y2, pwh3, mu2, rs2, b3, Kb);

  // PAC stage (WMMA + TDM K-tile load)
  pac_wmma_kernel<<<segBlocks, 512, 0, stream>>>(x, Kb, pwhP, pacb, out);
}
